// GraphAttentionLayer_9079560864453
// MI455X (gfx1250) — matrix operand order repaired and checked by reading, not compiled, not run on hardware
//
#include <hip/hip_runtime.h>
#include <hip/hip_bf16.h>

#define NN     8192
#define IN_F   512
#define OUT_F  256
#define ALPHA  0.2f
#define MASKV  (-9e15f)

typedef __attribute__((ext_vector_type(16))) __bf16 v16bf;
typedef __attribute__((ext_vector_type(8)))  float  v8f;

union BFrag {
    unsigned short us[16];
    unsigned int   u32[8];
    v16bf          v;
};

// f32 -> bf16 (native conversion; backend picks hw cvt when available)
static __device__ __forceinline__ unsigned short f2bf(float f) {
    __bf16 b = (__bf16)f;
    unsigned short u;
    __builtin_memcpy(&u, &b, sizeof(u));
    return u;
}

static __device__ __forceinline__ float lrelu(float v) {
    return v >= 0.f ? v : ALPHA * v;
}

// async DMA: 16 bytes global -> LDS, tracked by ASYNCcnt
static __device__ __forceinline__ void async_b128(unsigned lds_off,
                                                  const void* gptr) {
    unsigned long long ga = (unsigned long long)gptr;
    asm volatile("global_load_async_to_lds_b128 %0, %1, off"
                 :: "v"(lds_off), "v"(ga) : "memory");
}

// ---------------------------------------------------------------------------
// Kernel 1: h = X@W (f32 + packed bf16), hc = X@W_gc (packed bf16).
// Packed layout: Pk[(k>>1)*OUT_F + n] = {bf16 h[k][n] (lo), bf16 h[k+1][n] (hi)}
// ---------------------------------------------------------------------------
__global__ __launch_bounds__(256) void gat_gemm_in(
    const float* __restrict__ X, const float* __restrict__ W0,
    const float* __restrict__ W1, float* __restrict__ Hf,
    unsigned int* __restrict__ Hpk, unsigned int* __restrict__ HCpk)
{
    const int mt    = blockIdx.x;
    const int which = blockIdx.y;
    const float*  Wm = which ? W1  : W0;
    unsigned int* Pk = which ? HCpk : Hpk;

    const int t    = threadIdx.x;
    const int lane = t & 31;
    const int wave = t >> 5;
    const int hi   = lane >> 4;
    const int l16  = lane & 15;
    const int gm   = mt * 16;

    __shared__ unsigned short sA[16 * 32];
    const unsigned int* sA32 = (const unsigned int*)sA;

    const int prow = t >> 4;
    const int pk2  = (t & 15) * 2;

    const int n0 = wave * 32;
    const int c0 = n0 + l16;
    const int c1 = n0 + 16 + l16;

    v8f acc0 = {}; v8f acc1 = {};

    for (int k0 = 0; k0 < IN_F; k0 += 32) {
        const float* xr = X + (size_t)(gm + prow) * IN_F + k0 + pk2;
        sA[prow * 32 + pk2]     = f2bf(xr[0]);
        sA[prow * 32 + pk2 + 1] = f2bf(xr[1]);
        __syncthreads();

        BFrag a;
#pragma unroll
        for (int j = 0; j < 8; ++j) {
            const int kb = (hi ? 16 : 0) + j * 2;
            a.u32[j] = sA32[(l16 * 32 + kb) >> 1];
        }
        const int kb2 = k0 + (hi ? 16 : 0);
        BFrag b0, b1;
#pragma unroll
        for (int j = 0; j < 8; ++j) {
            b0.us[2*j]   = f2bf(Wm[(size_t)(kb2 + 2*j    ) * OUT_F + c0]);
            b0.us[2*j+1] = f2bf(Wm[(size_t)(kb2 + 2*j + 1) * OUT_F + c0]);
            b1.us[2*j]   = f2bf(Wm[(size_t)(kb2 + 2*j    ) * OUT_F + c1]);
            b1.us[2*j+1] = f2bf(Wm[(size_t)(kb2 + 2*j + 1) * OUT_F + c1]);
        }
        acc0 = __builtin_amdgcn_wmma_f32_16x16x32_bf16(false, a.v, false, b0.v,
                                                       (short)0, acc0, false, false);
        acc1 = __builtin_amdgcn_wmma_f32_16x16x32_bf16(false, a.v, false, b1.v,
                                                       (short)0, acc1, false, false);
        __syncthreads();
    }

    const int rbase = gm + (hi ? 8 : 0);
#pragma unroll
    for (int j = 0; j < 8; j += 2) {
        const unsigned int p0 =
            (unsigned int)f2bf(acc0[j]) | ((unsigned int)f2bf(acc0[j+1]) << 16);
        const unsigned int p1 =
            (unsigned int)f2bf(acc1[j]) | ((unsigned int)f2bf(acc1[j+1]) << 16);
        const size_t ppix = (size_t)((rbase + j) >> 1) * OUT_F;
        Pk[ppix + c0] = p0;
        Pk[ppix + c1] = p1;
    }
    if (which == 0) {
#pragma unroll
        for (int j = 0; j < 8; ++j) {
            Hf[(size_t)(rbase + j) * OUT_F + c0] = acc0[j];
            Hf[(size_t)(rbase + j) * OUT_F + c1] = acc1[j];
        }
    }
}

// ---------------------------------------------------------------------------
// Kernel 2: f_src[i] = h[i]·a[0:256], f_dst[i] = h[i]·a[256:512]
// ---------------------------------------------------------------------------
__global__ __launch_bounds__(256) void gat_fdot(
    const float* __restrict__ Hf, const float* __restrict__ avec,
    float* __restrict__ fsrc, float* __restrict__ fdst)
{
    const int i = blockIdx.x;
    const int t = threadIdx.x;
    __shared__ float r1[256];
    __shared__ float r2[256];
    const float hv = Hf[(size_t)i * OUT_F + t];
    r1[t] = hv * avec[t];
    r2[t] = hv * avec[OUT_F + t];
    __syncthreads();
    for (int s = 128; s > 0; s >>= 1) {
        if (t < s) { r1[t] += r1[t + s]; r2[t] += r2[t + s]; }
        __syncthreads();
    }
    if (t == 0) { fsrc[i] = r1[0]; fdst[i] = r2[0]; }
}

// ---------------------------------------------------------------------------
// Kernel 3: per-row streaming softmax stats (single pass over adj row).
// ---------------------------------------------------------------------------
__global__ __launch_bounds__(256) void gat_rowstats(
    const float* __restrict__ adj, const float* __restrict__ fsrc,
    const float* __restrict__ fdst, float* __restrict__ rowm,
    float* __restrict__ riz)
{
    const int i = blockIdx.x;
    const int t = threadIdx.x;
    const float fs = fsrc[i];
    const float* arow = adj + (size_t)i * NN;

    float m = -INFINITY, s = 0.f;
    for (int c = t; c < NN; c += 256) {
        const float a = arow[c];
        const float v = (a > 0.f) ? lrelu(fs + fdst[c]) : MASKV;
        if (v > m) { s = s * __expf(m - v) + 1.f; m = v; }
        else       { s += __expf(v - m); }
    }
    __shared__ float sm[256];
    __shared__ float ss[256];
    sm[t] = m; ss[t] = s;
    __syncthreads();
    for (int st = 128; st > 0; st >>= 1) {
        if (t < st) {
            const float m2 = sm[t + st], s2 = ss[t + st];
            const float M  = fmaxf(sm[t], m2);
            ss[t] = ss[t] * __expf(sm[t] - M) + s2 * __expf(m2 - M);
            sm[t] = M;
        }
        __syncthreads();
    }
    if (t == 0) { rowm[i] = sm[0]; riz[i] = 1.f / ss[0]; }
}

// ---------------------------------------------------------------------------
// Kernel 4: fused out = softmax(att)@h + adj@hc + b_gc.
// 32-row tile per block. adj tiles (32x32 f32 = 4 KB) are streamed into LDS
// with double-buffered global_load_async_to_lds_b128 (ASYNCcnt), transformed
// in place to bf16 att/adj A-tiles, then 8 waves x (2 M x 2 N) issue
// 8 v_wmma_f32_16x16x32_bf16 per K-step, sharing accumulators for both terms.
// ---------------------------------------------------------------------------
__global__ __launch_bounds__(256) void gat_aggregate(
    const float* __restrict__ adj, const float* __restrict__ fsrc,
    const float* __restrict__ fdst, const float* __restrict__ rowm,
    const float* __restrict__ riz, const unsigned int* __restrict__ Hpk,
    const unsigned int* __restrict__ HCpk, const float* __restrict__ bgc,
    float* __restrict__ out)
{
    const int gm   = blockIdx.x * 32;
    const int t    = threadIdx.x;
    const int lane = t & 31;
    const int wave = t >> 5;
    const int hi   = lane >> 4;
    const int l16  = lane & 15;

    __shared__ float          sRaw[2][32 * 32];   // raw f32 adj tiles (DMA dest)
    __shared__ unsigned short sAtt[32 * 32];      // bf16 attention tile
    __shared__ unsigned short sAdj[32 * 32];      // bf16 adjacency tile
    const unsigned int* sAtt32 = (const unsigned int*)sAtt;
    const unsigned int* sAdj32 = (const unsigned int*)sAdj;
    unsigned int* wAtt32 = (unsigned int*)sAtt;
    unsigned int* wAdj32 = (unsigned int*)sAdj;

    const int   prow = t >> 3;          // 0..31 : row this thread preps
    const int   pseg = (t & 7) * 4;     // 0,4,...,28 : 4-float segment
    const float fs   = fsrc[gm + prow];
    const float m    = rowm[gm + prow];
    const float iz   = riz [gm + prow];
    const float* arow = adj + (size_t)(gm + prow) * NN + pseg;

    const unsigned ldsRaw0 = (unsigned)(size_t)&sRaw[0][prow * 32 + pseg];
    const unsigned ldsRaw1 = (unsigned)(size_t)&sRaw[1][prow * 32 + pseg];

    const int n0 = wave * 32;
    const int c0 = n0 + l16;
    const int c1 = n0 + 16 + l16;

    v8f acc00 = {}, acc01 = {}, acc10 = {}, acc11 = {};

    // prologue: DMA first tile into buffer 0
    async_b128(ldsRaw0, arow);

    int buf = 0;
    for (int k0 = 0; k0 < NN; k0 += 32) {
        const bool more = (k0 + 32) < NN;
        if (more) {
            async_b128(buf ? ldsRaw0 : ldsRaw1, arow + k0 + 32);
            asm volatile("s_wait_asynccnt 0x1" ::: "memory");  // current tile done
        } else {
            asm volatile("s_wait_asynccnt 0x0" ::: "memory");
        }
        __syncthreads();   // A: tile visible workgroup-wide

        // transform raw f32 adjacency -> bf16 att / adj tiles
        const float4 a4 = *(const float4*)&sRaw[buf][prow * 32 + pseg];
        const float v0 = (a4.x > 0.f) ? lrelu(fs + fdst[k0 + pseg + 0]) : MASKV;
        const float v1 = (a4.y > 0.f) ? lrelu(fs + fdst[k0 + pseg + 1]) : MASKV;
        const float v2 = (a4.z > 0.f) ? lrelu(fs + fdst[k0 + pseg + 2]) : MASKV;
        const float v3 = (a4.w > 0.f) ? lrelu(fs + fdst[k0 + pseg + 3]) : MASKV;
        const unsigned base = (unsigned)(prow * 32 + pseg) >> 1;
        wAtt32[base]     = (unsigned)f2bf(__expf(v0 - m) * iz) |
                           ((unsigned)f2bf(__expf(v1 - m) * iz) << 16);
        wAtt32[base + 1] = (unsigned)f2bf(__expf(v2 - m) * iz) |
                           ((unsigned)f2bf(__expf(v3 - m) * iz) << 16);
        wAdj32[base]     = (unsigned)f2bf(a4.x) | ((unsigned)f2bf(a4.y) << 16);
        wAdj32[base + 1] = (unsigned)f2bf(a4.z) | ((unsigned)f2bf(a4.w) << 16);
        __syncthreads();   // B: bf16 tiles ready; also fences raw-buffer reuse

        // A fragments (two 16-row M-tiles) from LDS
        BFrag aa0, ag0, aa1, ag1;
#pragma unroll
        for (int j = 0; j < 8; ++j) {
            const int kb = (hi ? 16 : 0) + j * 2;
            const int r0 = (l16) * 32 + kb;
            const int r1 = (16 + l16) * 32 + kb;
            aa0.u32[j] = sAtt32[r0 >> 1];
            ag0.u32[j] = sAdj32[r0 >> 1];
            aa1.u32[j] = sAtt32[r1 >> 1];
            ag1.u32[j] = sAdj32[r1 >> 1];
        }
        // B fragments (K-pair packed h / hc), reused across both M-tiles
        const int kp = (k0 + (hi ? 16 : 0)) >> 1;
        BFrag bh0, bc0, bh1, bc1;
#pragma unroll
        for (int j = 0; j < 8; ++j) {
            bh0.u32[j] = Hpk [(size_t)(kp + j) * OUT_F + c0];
            bc0.u32[j] = HCpk[(size_t)(kp + j) * OUT_F + c0];
            bh1.u32[j] = Hpk [(size_t)(kp + j) * OUT_F + c1];
            bc1.u32[j] = HCpk[(size_t)(kp + j) * OUT_F + c1];
        }
        acc00 = __builtin_amdgcn_wmma_f32_16x16x32_bf16(false, aa0.v, false, bh0.v,
                                                        (short)0, acc00, false, false);
        acc00 = __builtin_amdgcn_wmma_f32_16x16x32_bf16(false, ag0.v, false, bc0.v,
                                                        (short)0, acc00, false, false);
        acc01 = __builtin_amdgcn_wmma_f32_16x16x32_bf16(false, aa0.v, false, bh1.v,
                                                        (short)0, acc01, false, false);
        acc01 = __builtin_amdgcn_wmma_f32_16x16x32_bf16(false, ag0.v, false, bc1.v,
                                                        (short)0, acc01, false, false);
        acc10 = __builtin_amdgcn_wmma_f32_16x16x32_bf16(false, aa1.v, false, bh0.v,
                                                        (short)0, acc10, false, false);
        acc10 = __builtin_amdgcn_wmma_f32_16x16x32_bf16(false, ag1.v, false, bc0.v,
                                                        (short)0, acc10, false, false);
        acc11 = __builtin_amdgcn_wmma_f32_16x16x32_bf16(false, aa1.v, false, bh1.v,
                                                        (short)0, acc11, false, false);
        acc11 = __builtin_amdgcn_wmma_f32_16x16x32_bf16(false, ag1.v, false, bc1.v,
                                                        (short)0, acc11, false, false);
        buf ^= 1;
    }

    const float bg0 = bgc[c0];
    const float bg1 = bgc[c1];
    const int r0 = gm + (hi ? 8 : 0);
    const int r1 = gm + 16 + (hi ? 8 : 0);
#pragma unroll
    for (int j = 0; j < 8; ++j) {
        out[(size_t)(r0 + j) * OUT_F + c0] = acc00[j] + bg0;
        out[(size_t)(r0 + j) * OUT_F + c1] = acc01[j] + bg1;
        out[(size_t)(r1 + j) * OUT_F + c0] = acc10[j] + bg0;
        out[(size_t)(r1 + j) * OUT_F + c1] = acc11[j] + bg1;
    }
}

// ---------------------------------------------------------------------------
extern "C" void kernel_launch(void* const* d_in, const int* in_sizes, int n_in,
                              void* d_out, int out_size, void* d_ws, size_t ws_size,
                              hipStream_t stream) {
    const float* X    = (const float*)d_in[0];   // [8192, 512]
    const float* ADJ  = (const float*)d_in[1];   // [8192, 8192]
    const float* W    = (const float*)d_in[2];   // [512, 256]
    const float* AV   = (const float*)d_in[3];   // [512, 1]
    const float* WGC  = (const float*)d_in[4];   // [512, 256]
    const float* BGC  = (const float*)d_in[5];   // [256]
    float* OUT = (float*)d_out;                  // [8192, 256]

    char* p = (char*)d_ws;
    float*        Hf   = (float*)p;        p += (size_t)NN * OUT_F * 4;       // 8 MB
    unsigned int* Hpk  = (unsigned int*)p; p += (size_t)(NN/2) * OUT_F * 4;   // 4 MB
    unsigned int* HCpk = (unsigned int*)p; p += (size_t)(NN/2) * OUT_F * 4;   // 4 MB
    float*        fsrc = (float*)p;        p += (size_t)NN * 4;
    float*        fdst = (float*)p;        p += (size_t)NN * 4;
    float*        rowm = (float*)p;        p += (size_t)NN * 4;
    float*        riz  = (float*)p;        p += (size_t)NN * 4;

    gat_gemm_in<<<dim3(NN / 16, 2), 256, 0, stream>>>(X, W, WGC, Hf, Hpk, HCpk);
    gat_fdot<<<NN, 256, 0, stream>>>(Hf, AV, fsrc, fdst);
    gat_rowstats<<<NN, 256, 0, stream>>>(ADJ, fsrc, fdst, rowm, riz);
    gat_aggregate<<<NN / 32, 256, 0, stream>>>(ADJ, fsrc, fdst, rowm, riz,
                                               Hpk, HCpk, BGC, OUT);
}
